// Attention_11768210391583
// MI455X (gfx1250) — matrix operand order repaired and checked by reading, not compiled, not run on hardware
//
#include <hip/hip_runtime.h>

#define B_   8
#define N_   1024
#define C_   768
#define H_   12
#define HD_  64
#define SCALE_ 0.125f   // HD^-0.5

typedef __attribute__((ext_vector_type(16))) _Float16 v16h;
typedef __attribute__((ext_vector_type(8)))  _Float16 v8h;
typedef __attribute__((ext_vector_type(8)))  float    v8f;

static __device__ __forceinline__ v8f vzero8() {
  v8f z;
#pragma unroll
  for (int i = 0; i < 8; ++i) z[i] = 0.0f;
  return z;
}

// Build a 16-half fragment from two 16-byte chunks.
static __device__ __forceinline__ v16h frag16(const _Float16* p0, const _Float16* p1) {
  v8h lo = *(const v8h*)p0;
  v8h hi = *(const v8h*)p1;
  v16h r;
#pragma unroll
  for (int i = 0; i < 8; ++i) { r[i] = lo[i]; r[i + 8] = hi[i]; }
  return r;
}

static __device__ __forceinline__ v8f wmma_f16(v16h a, v16h b, v8f c) {
  return __builtin_amdgcn_wmma_f32_16x16x32_f16(false, a, false, b, (short)0, c,
                                                false, false);
}

// ---------------------------------------------------------------------------
// Async global -> LDS 16-byte copy (GLOBAL_LOAD_ASYNC_TO_LDS_B128, ASYNCcnt).
// VFLAT async form: "vdst(LDS byte addr), vaddr(64b global), off". Flat
// pointers to LDS carry the LDS byte offset in their low 32 bits (aperture
// check truncates to ADDR[31:0]), so the LDS VGPR is the truncated pointer.
// ---------------------------------------------------------------------------
static __device__ __forceinline__ void async_b128(const _Float16* g, _Float16* l) {
  unsigned loff = (unsigned)(uintptr_t)l;             // LDS byte address
  unsigned long long ga = (unsigned long long)(uintptr_t)g;
  asm volatile("global_load_async_to_lds_b128 %0, %1, off"
               :: "v"(loff), "v"(ga) : "memory");
}

#define WAIT_ASYNC(n) asm volatile("s_wait_asynccnt %0" :: "n"(n) : "memory")

// ---------------------------------------------------------------------------
// fp32 -> f16 convert
// ---------------------------------------------------------------------------
__global__ void cvt_f32_f16(const float* __restrict__ in, _Float16* __restrict__ out,
                            int n) {
  int i = blockIdx.x * blockDim.x + threadIdx.x;
  int stride = gridDim.x * blockDim.x;
  for (; i < n; i += stride) out[i] = (_Float16)in[i];
}

// ---------------------------------------------------------------------------
// Block-level GEMM mainloop with async-LDS double buffering.
// 256 threads = 8 waves compute a 128(M) x 64(N) macro-tile; wave w owns the
// 32x32 tile at (mw = (w>>1)*32, nw = (w&1)*32); acc = 4 16x16 sub-tiles.
// A, B row-major f16, K multiple of 32.
// A-frag lane pattern: lane<16 K{0..15};       lane>=16 K{16..31}
// B-frag lane pattern: lane<16 K{0..15};       lane>=16 K{16..31}
// ---------------------------------------------------------------------------
static __device__ __forceinline__ void gemm_block(const _Float16* __restrict__ A,
                                                  const _Float16* __restrict__ Bm,
                                                  int K, int mBase, int nBase,
                                                  v8f acc[4]) {
  __shared__ alignas(16) _Float16 sA[2][128][32];   // 16 KB
  __shared__ alignas(16) _Float16 sB[2][64][32];    //  8 KB

  const int t    = threadIdx.x;
  const int lane = t & 31;
  const int wave = t >> 5;
  const int mw   = (wave >> 1) * 32;
  const int nw   = (wave & 1) * 32;
  const int r    = lane & 15;
  const int ka   = (lane >= 16) ? 8 : 0;
  const int kb   = (lane >= 16) ? 16 : 0;

  // Branch-free staging of one K=32 slab: 3 x 16B per lane covers
  // A rows {t>>2, t>>2+64} and B row {t>>2} (4 chunks per row, 256 lanes).
  const int srow = t >> 2;
  const int sch  = (t & 3) << 3;
  const _Float16* gA0 = A  + (size_t)(mBase + srow)      * K + sch;
  const _Float16* gA1 = A  + (size_t)(mBase + 64 + srow) * K + sch;
  const _Float16* gB  = Bm + (size_t)(nBase + srow)      * K + sch;

  auto stage = [&](int buf, int k0) {
    async_b128(gA0 + k0, &sA[buf][srow][sch]);
    async_b128(gA1 + k0, &sA[buf][srow + 64][sch]);
    async_b128(gB  + k0, &sB[buf][srow][sch]);
  };

  auto compute = [&](int buf) {
    v16h a0 = frag16(&sA[buf][mw + r][kb],      &sA[buf][mw + r][kb + 8]);
    v16h a1 = frag16(&sA[buf][mw + 16 + r][kb], &sA[buf][mw + 16 + r][kb + 8]);
    v16h b0 = frag16(&sB[buf][nw + r][kb],      &sB[buf][nw + r][kb + 8]);
    v16h b1 = frag16(&sB[buf][nw + 16 + r][kb], &sB[buf][nw + 16 + r][kb + 8]);
    acc[0] = wmma_f16(a0, b0, acc[0]);
    acc[1] = wmma_f16(a0, b1, acc[1]);
    acc[2] = wmma_f16(a1, b0, acc[2]);
    acc[3] = wmma_f16(a1, b1, acc[3]);
  };

  int buf = 0;
  stage(0, 0);
  for (int k0 = 32; k0 < K; k0 += 32) {   // straight-line steady state
    stage(buf ^ 1, k0);     // prefetch next slab
    WAIT_ASYNC(3);          // retire current slab (in-order completion)
    __syncthreads();        // slab visible to all waves
    compute(buf);
    __syncthreads();        // all reads done before slab is overwritten
    buf ^= 1;
  }
  WAIT_ASYNC(0);            // peeled last slab
  __syncthreads();
  compute(buf);
}

// ---------------------------------------------------------------------------
// QKV projection: [8192 x 768] x [2304 x 768]^T + bias, fused epilogue that
// scatters q (pre-scaled), k into [B,H,N,HD] f16 and v transposed [B,H,HD,N].
// Grid: (8192/128) * (2304/64) = 64 * 36 = 2304 blocks.
// ---------------------------------------------------------------------------
__global__ void gemm_qkv(const _Float16* __restrict__ xh,
                         const _Float16* __restrict__ wh,
                         const float* __restrict__ bias,
                         _Float16* __restrict__ qh, _Float16* __restrict__ kh,
                         _Float16* __restrict__ vth) {
  const int ntn = (3 * C_) / 64;            // 36 col macro-tiles
  int mBase = (blockIdx.x / ntn) * 128;
  int nBase = (blockIdx.x % ntn) * 64;
  int lane = threadIdx.x & 31;
  int wave = threadIdx.x >> 5;
  int mw = (wave >> 1) * 32, nw = (wave & 1) * 32;

  v8f acc[4] = {vzero8(), vzero8(), vzero8(), vzero8()};
  gemm_block(xh, wh, C_, mBase, nBase, acc);

  int hi8 = (lane >= 16) ? 8 : 0;
  int cn  = lane & 15;
#pragma unroll
  for (int tt = 0; tt < 4; ++tt) {
    int mm = mBase + mw + (tt >> 1) * 16 + hi8;
    int oo = nBase + nw + (tt & 1) * 16 + cn;   // output channel in [0, 3C)
    int sel = oo / C_;                          // 0=q 1=k 2=v (uniform per subtile)
    int rem = oo - sel * C_;
    int hh  = rem / HD_;
    int d   = rem - hh * HD_;
    float bz = bias[oo];
#pragma unroll
    for (int rr = 0; rr < 8; ++rr) {
      int m = mm + rr;
      int b = m >> 10;                          // N_ == 1024
      int n = m & (N_ - 1);
      float v = acc[tt][rr] + bz;
      size_t bhn = ((size_t)(b * H_ + hh)) * N_;
      if (sel == 0) {
        qh[(bhn + n) * HD_ + d] = (_Float16)(v * SCALE_);
      } else if (sel == 1) {
        kh[(bhn + n) * HD_ + d] = (_Float16)v;
      } else {
        vth[((size_t)(b * H_ + hh) * HD_ + d) * N_ + n] = (_Float16)v;
      }
    }
  }
}

// ---------------------------------------------------------------------------
// Output projection: [8192 x 768] x [768 x 768]^T + bias -> fp32 out.
// Grid: 64 * (768/64) = 768 blocks.
// ---------------------------------------------------------------------------
__global__ void gemm_proj(const _Float16* __restrict__ ah,
                          const _Float16* __restrict__ wh,
                          const float* __restrict__ bias,
                          float* __restrict__ out) {
  const int ntn = C_ / 64;                  // 12 col macro-tiles
  int mBase = (blockIdx.x / ntn) * 128;
  int nBase = (blockIdx.x % ntn) * 64;
  int lane = threadIdx.x & 31;
  int wave = threadIdx.x >> 5;
  int mw = (wave >> 1) * 32, nw = (wave & 1) * 32;

  v8f acc[4] = {vzero8(), vzero8(), vzero8(), vzero8()};
  gemm_block(ah, wh, C_, mBase, nBase, acc);

  int hi8 = (lane >= 16) ? 8 : 0;
  int cn  = lane & 15;
#pragma unroll
  for (int tt = 0; tt < 4; ++tt) {
    int mm = mBase + mw + (tt >> 1) * 16 + hi8;
    int oo = nBase + nw + (tt & 1) * 16 + cn;
    float bz = bias[oo];
#pragma unroll
    for (int rr = 0; rr < 8; ++rr)
      out[(size_t)(mm + rr) * C_ + oo] = acc[tt][rr] + bz;
  }
}

// ---------------------------------------------------------------------------
// Flash attention, causal. One 32-thread block (single wave) owns 16 query
// rows of one (b,h). Q pre-scaled by 1/sqrt(HD). KV processed in 32-row
// blocks (= WMMA K) with online softmax; P bounced through 1KB LDS to
// re-layout C-fragments into an A-fragment.
// ---------------------------------------------------------------------------
__global__ void flash_attn(const _Float16* __restrict__ qh,
                           const _Float16* __restrict__ kh,
                           const _Float16* __restrict__ vth,
                           _Float16* __restrict__ aoh) {
  __shared__ alignas(16) _Float16 lds[16][32];

  const int qblocks = N_ / 16;              // 64
  int bh = blockIdx.x / qblocks;
  int qb = blockIdx.x % qblocks;
  int b = bh / H_, h = bh % H_;
  int m0 = qb * 16;

  int lane = threadIdx.x & 31;
  int r    = lane & 15;
  int hi   = (lane >= 16);
  int hi8  = hi ? 8 : 0;
  int ka   = hi ? 8 : 0;                    // unused; A-frags also use kb
  int kb   = hi ? 16 : 0;                   // B-frag K offset

  const _Float16* Q = qh  + (size_t)bh * N_ * HD_;
  const _Float16* K = kh  + (size_t)bh * N_ * HD_;
  const _Float16* V = vth + (size_t)bh * HD_ * N_;

  // Q fragments (rows m0..m0+15, d 0..31 and 32..63)
  const _Float16* qp = Q + (size_t)(m0 + r) * HD_;
  v16h aq0 = frag16(qp + kb,      qp + kb + 8);
  v16h aq1 = frag16(qp + 32 + kb, qp + 32 + kb + 8);

  float mi[8], li[8];
  v8f o[4];
#pragma unroll
  for (int e = 0; e < 8; ++e) { mi[e] = -__builtin_inff(); li[e] = 0.0f; }
#pragma unroll
  for (int tt = 0; tt < 4; ++tt) o[tt] = vzero8();

  for (int j0 = 0; j0 < m0 + 16; j0 += 32) {
    // ---- S = Q * K^T over this 32-col KV block (two 16-col tiles) ----
    v8f s[2];
#pragma unroll
    for (int u = 0; u < 2; ++u) {
      const _Float16* kp = K + (size_t)(j0 + u * 16 + r) * HD_;
      v16h bk0 = frag16(kp + kb,      kp + kb + 8);        // d 0..31
      v16h bk1 = frag16(kp + 32 + kb, kp + 32 + kb + 8);   // d 32..63
      v8f z = vzero8();
      z = wmma_f16(aq0, bk0, z);
      z = wmma_f16(aq1, bk1, z);
      s[u] = z;
    }

    // ---- causal mask + online softmax stats (per output row e) ----
    float alpha[8];
#pragma unroll
    for (int e = 0; e < 8; ++e) {
      int m  = m0 + e + hi8;                // global query row
      int jc = j0 + (lane & 15);            // key col of tile 0
      float s0 = (jc      <= m) ? s[0][e] : -__builtin_inff();
      float s1 = (jc + 16 <= m) ? s[1][e] : -__builtin_inff();
      float mx = fmaxf(s0, s1);
#pragma unroll
      for (int off = 1; off < 16; off <<= 1)
        mx = fmaxf(mx, __shfl_xor(mx, off, 32));
      float mn = fmaxf(mi[e], mx);          // finite after first block (col 0)
      float al = __expf(mi[e] - mn);
      float e0 = __expf(s0 - mn);
      float e1 = __expf(s1 - mn);
      float rs = e0 + e1;
#pragma unroll
      for (int off = 1; off < 16; off <<= 1)
        rs += __shfl_xor(rs, off, 32);
      li[e]    = li[e] * al + rs;
      mi[e]    = mn;
      alpha[e] = al;
      // stage P (f16) row-major in LDS for the A-fragment re-layout
      lds[e + hi8][(lane & 15)]      = (_Float16)e0;
      lds[e + hi8][16 + (lane & 15)] = (_Float16)e1;
    }

    // rescale accumulators by alpha (row-replicated across the lane half)
#pragma unroll
    for (int tt = 0; tt < 4; ++tt)
#pragma unroll
      for (int e = 0; e < 8; ++e) o[tt][e] *= alpha[e];

    __syncthreads();                        // single-wave WG: cheap, orders LDS
    v16h ap = frag16(&lds[r][kb], &lds[r][kb + 8]);
    __syncthreads();                        // keep next iter's stores after load

    // ---- O += P * V  (V pre-transposed: row d, col n) ----
#pragma unroll
    for (int tt = 0; tt < 4; ++tt) {
      const _Float16* vp = V + (size_t)(tt * 16 + r) * N_ + j0 + kb;
      v16h bv = frag16(vp, vp + 8);
      o[tt] = wmma_f16(ap, bv, o[tt]);
    }
  }

  // ---- normalize and store to [B, N, C] f16 (C index = h*HD + d) ----
  _Float16* outp = aoh + (size_t)b * N_ * C_ + (size_t)h * HD_;
#pragma unroll
  for (int tt = 0; tt < 4; ++tt) {
#pragma unroll
    for (int e = 0; e < 8; ++e) {
      int m = m0 + e + hi8;
      int d = tt * 16 + (lane & 15);
      outp[(size_t)m * C_ + d] = (_Float16)(o[tt][e] / li[e]);
    }
  }
}

// ---------------------------------------------------------------------------
// Host-side launcher
// ---------------------------------------------------------------------------
extern "C" void kernel_launch(void* const* d_in, const int* in_sizes, int n_in,
                              void* d_out, int out_size, void* d_ws, size_t ws_size,
                              hipStream_t stream) {
  (void)in_sizes; (void)n_in; (void)out_size; (void)ws_size;
  const float* x      = (const float*)d_in[0];
  // d_in[1] = attn_mask (bool causal) -- handled analytically in flash_attn
  const float* qkv_w  = (const float*)d_in[2];
  const float* qkv_b  = (const float*)d_in[3];
  const float* proj_w = (const float*)d_in[4];
  const float* proj_b = (const float*)d_in[5];
  float* out = (float*)d_out;

  char* ws = (char*)d_ws;
  size_t off = 0;
  auto alloc = [&](size_t bytes) {
    char* p = ws + off;
    off += (bytes + 255) & ~(size_t)255;
    return p;
  };
  _Float16* xh  = (_Float16*)alloc((size_t)B_ * N_ * C_ * 2);
  _Float16* qwh = (_Float16*)alloc((size_t)3 * C_ * C_ * 2);
  _Float16* pwh = (_Float16*)alloc((size_t)C_ * C_ * 2);
  _Float16* qh  = (_Float16*)alloc((size_t)B_ * H_ * N_ * HD_ * 2);
  _Float16* kh  = (_Float16*)alloc((size_t)B_ * H_ * N_ * HD_ * 2);
  _Float16* vth = (_Float16*)alloc((size_t)B_ * H_ * N_ * HD_ * 2);
  _Float16* aoh = (_Float16*)alloc((size_t)B_ * N_ * C_ * 2);

  cvt_f32_f16<<<1024, 256, 0, stream>>>(x, xh, B_ * N_ * C_);
  cvt_f32_f16<<<1024, 256, 0, stream>>>(qkv_w, qwh, 3 * C_ * C_);
  cvt_f32_f16<<<512, 256, 0, stream>>>(proj_w, pwh, C_ * C_);

  // 128x64 macro-tiles: 64*36 = 2304 blocks
  gemm_qkv<<<2304, 256, 0, stream>>>(xh, qwh, qkv_b, qh, kh, vth);

  // one single-wave block per (b, h, 16-row query block)
  flash_attn<<<B_ * H_ * (N_ / 16), 32, 0, stream>>>(qh, kh, vth, aoh);

  // 128x64 macro-tiles: 64*12 = 768 blocks
  gemm_proj<<<768, 256, 0, stream>>>(aoh, pwh, proj_b, out);
}
